// Model_29515015258437
// MI455X (gfx1250) — compile-verified, matrix operand order checked by reading, not run on hardware
//
#include <hip/hip_runtime.h>
#include <hip/hip_bf16.h>
#include <math.h>

// ---------- types for WMMA ----------
typedef __attribute__((ext_vector_type(8)))  __bf16 v8bf;
typedef __attribute__((ext_vector_type(16))) __bf16 v16bf;
typedef __attribute__((ext_vector_type(8)))  float  v8f;

// =====================================================================
// fp32 -> bf16 (round-to-nearest-even), stored as ushort (same bits)
// =====================================================================
__global__ void k_f32_to_bf16(const float* __restrict__ src,
                              unsigned short* __restrict__ dst, size_t n) {
  size_t i = (size_t)blockIdx.x * blockDim.x + threadIdx.x;
  if (i >= n) return;
  unsigned u = __float_as_uint(src[i]);
  u = (u + 0x7FFFu + ((u >> 16) & 1u)) >> 16;
  dst[i] = (unsigned short)u;
}

// bias_comb = a + b  (tiny vector)
__global__ void k_add_vec(const float* __restrict__ a, const float* __restrict__ b,
                          float* __restrict__ c, int n) {
  int i = blockIdx.x * blockDim.x + threadIdx.x;
  if (i < n) c[i] = a[i] + b[i];
}

__device__ __forceinline__ v16bf ld_frag(const __bf16* p, int k) {
  v8bf lo = *(const v8bf*)(p + k);
  v8bf hi = *(const v8bf*)(p + k + 16);
  return __builtin_shufflevector(lo, hi, 0,1,2,3,4,5,6,7,8,9,10,11,12,13,14,15);
}

// =====================================================================
// bf16 WMMA GEMM:  C[M,Nc] = A[M,K] @ B[Nc,K]^T  (+ bias[col])
// Register-blocked: each wave computes a 16x64 C tile = 4 accumulators
// along N sharing ONE A fragment (A loads amortized 4x; 4 WMMAs / k-step).
// 4 waves/block -> block covers 16 rows x 256 cols.
// Per-lane fragment layout per CDNA5 ISA (16-bit A 16x32 / B 32x16):
//   lane l: row/col = l&15, half = l>>4; 16 bf16 = two contiguous 16B
//   chunks at k + 8*half and k + 16 + 8*half.
// =====================================================================
__global__ __launch_bounds__(128)
void k_gemm_bf16_wmma(const __bf16* __restrict__ A, const __bf16* __restrict__ B,
                      const float* __restrict__ bias, float* __restrict__ C,
                      int M, int Nc, int K) {
  const int wave = threadIdx.x >> 5;
  const int lane = threadIdx.x & 31;
  const int n0 = (blockIdx.x * 4 + wave) * 64;   // 64 cols per wave
  const int m0 = blockIdx.y * 16;                // M = 50000 = 3125*16 exact
  if (n0 >= Nc) return;                          // wave-uniform: EXEC stays all-1
  const int r  = lane & 15;
  const int hi = lane >> 4;
  const __bf16* arow = A + (size_t)(m0 + r) * K + 8 * hi;
  // B row pointers for 4 column sub-tiles; clamp (not predicate) so all
  // lanes stay active for WMMA and no OOB access occurs.
  int nr0 = n0 +  0 + r; if (nr0 > Nc - 1) nr0 = Nc - 1;
  int nr1 = n0 + 16 + r; if (nr1 > Nc - 1) nr1 = Nc - 1;
  int nr2 = n0 + 32 + r; if (nr2 > Nc - 1) nr2 = Nc - 1;
  int nr3 = n0 + 48 + r; if (nr3 > Nc - 1) nr3 = Nc - 1;
  const __bf16* brow0 = B + (size_t)nr0 * K + 8 * hi;
  const __bf16* brow1 = B + (size_t)nr1 * K + 8 * hi;
  const __bf16* brow2 = B + (size_t)nr2 * K + 8 * hi;
  const __bf16* brow3 = B + (size_t)nr3 * K + 8 * hi;

  v8f acc0 = {}, acc1 = {}, acc2 = {}, acc3 = {};
#pragma unroll 2
  for (int k = 0; k < K; k += 32) {
    if (k + 32 < K) {                  // global_prefetch_b8 for next K-slab
      __builtin_prefetch(arow  + k + 32, 0, 1);
      __builtin_prefetch(brow0 + k + 32, 0, 1);
      __builtin_prefetch(brow2 + k + 32, 0, 1);
    }
    v16bf av  = ld_frag(arow,  k);
    v16bf bv0 = ld_frag(brow0, k);
    v16bf bv1 = ld_frag(brow1, k);
    v16bf bv2 = ld_frag(brow2, k);
    v16bf bv3 = ld_frag(brow3, k);
    acc0 = __builtin_amdgcn_wmma_f32_16x16x32_bf16(false, av, false, bv0, (short)0, acc0, false, false);
    acc1 = __builtin_amdgcn_wmma_f32_16x16x32_bf16(false, av, false, bv1, (short)0, acc1, false, false);
    acc2 = __builtin_amdgcn_wmma_f32_16x16x32_bf16(false, av, false, bv2, (short)0, acc2, false, false);
    acc3 = __builtin_amdgcn_wmma_f32_16x16x32_bf16(false, av, false, bv3, (short)0, acc3, false, false);
  }
  // C/D layout: VGPR i of lane l -> C[m0 + 8*(l>>4) + i, ncol + (l&15)]
  float* cbase = C + (size_t)(m0 + 8 * hi) * Nc + r;
#pragma unroll
  for (int j = 0; j < 4; ++j) {
    const int nc = n0 + j * 16;
    if (nc >= Nc) break;
    const v8f& a = j == 0 ? acc0 : j == 1 ? acc1 : j == 2 ? acc2 : acc3;
    const float bn = bias ? bias[nc + r] : 0.0f;
    float* crow = cbase + nc;
#pragma unroll
    for (int i = 0; i < 8; ++i) crow[(size_t)i * Nc] = a[i] + bn;
  }
}

// =====================================================================
// per-(node,head) attention scalars: asrc = <h[n,h,:], a_src[h,:]>, adst likewise
// one wave per (node,head), lane-parallel reduction
// =====================================================================
__global__ void k_node_attn(const float* __restrict__ hbuf,
                            const float* __restrict__ a_src,
                            const float* __restrict__ a_dst,
                            float* __restrict__ asrc, float* __restrict__ adst,
                            int N, int H, int D) {
  int w    = (int)(((size_t)blockIdx.x * blockDim.x + threadIdx.x) >> 5);
  int lane = threadIdx.x & 31;
  if (w >= N * H) return;
  int n = w / H, h = w - n * H;
  const float* hv = hbuf + (size_t)n * H * D + (size_t)h * D;
  const float* as = a_src + (size_t)h * D;
  const float* ad = a_dst + (size_t)h * D;
  float ss = 0.f, sd = 0.f;
  for (int d = lane; d < D; d += 32) { float v = hv[d]; ss += v * as[d]; sd += v * ad[d]; }
  for (int off = 16; off; off >>= 1) { ss += __shfl_xor(ss, off); sd += __shfl_xor(sd, off); }
  if (lane == 0) { asrc[w] = ss; adst[w] = sd; }
}

__global__ void k_init_softmax(float* __restrict__ mmax, float* __restrict__ denom, int n) {
  int i = blockIdx.x * blockDim.x + threadIdx.x;
  if (i < n) { mmax[i] = -__builtin_inff(); denom[i] = 0.f; }
}

__device__ __forceinline__ void atomicMaxFloat(float* addr, float val) {
  if (val >= 0.f) atomicMax((int*)addr, __float_as_int(val));
  else            atomicMin((unsigned int*)addr, __float_as_uint(val));
}

// pass1: alpha = leaky_relu(asrc[s]+adst[t]); segment max over dst
__global__ void k_edge_pass1(const long long* __restrict__ ei, int E, int N, int H,
                             const float* __restrict__ asrc, const float* __restrict__ adst,
                             float* __restrict__ alpha, float* __restrict__ mmax) {
  int idx = (int)((size_t)blockIdx.x * blockDim.x + threadIdx.x);
  int tot = (E + N) * H;
  if (idx >= tot) return;
  int e = idx / H, h = idx - e * H;
  int s, t;
  if (e < E) { s = (int)ei[e]; t = (int)ei[E + e]; } else { s = t = e - E; }
  float a = asrc[s * H + h] + adst[t * H + h];
  a = a > 0.f ? a : 0.2f * a;
  alpha[idx] = a;
  atomicMaxFloat(&mmax[t * H + h], a);
}

// pass2: e = exp(alpha - m); segment sum over dst
__global__ void k_edge_pass2(const long long* __restrict__ ei, int E, int N, int H,
                             float* __restrict__ alpha,
                             const float* __restrict__ mmax, float* __restrict__ denom) {
  int idx = (int)((size_t)blockIdx.x * blockDim.x + threadIdx.x);
  int tot = (E + N) * H;
  if (idx >= tot) return;
  int e = idx / H, h = idx - e * H;
  int t = (e < E) ? (int)ei[E + e] : (e - E);
  float m = mmax[t * H + h];
  if (!(m >= -3.0e38f)) m = 0.f;             // isfinite fallback (ref semantics)
  float ex = __expf(alpha[idx] - m);
  alpha[idx] = ex;
  atomicAdd(&denom[t * H + h], ex);
}

// pass3: acc[dst] += h[src] * (e / denom) * scale  — one wave per (edge, head)
__global__ void k_edge_msg(const long long* __restrict__ ei, int E, int N, int H, int D,
                           const float* __restrict__ alpha, const float* __restrict__ denom,
                           const float* __restrict__ hbuf, float* __restrict__ acc,
                           int concat, float scale) {
  int w    = (int)(((size_t)blockIdx.x * blockDim.x + threadIdx.x) >> 5);
  int lane = threadIdx.x & 31;
  int tot = (E + N) * H;
  if (w >= tot) return;
  int e = w / H, h = w - e * H;
  int s, t;
  if (e < E) { s = (int)ei[e]; t = (int)ei[E + e]; } else { s = t = e - E; }
  float wgt = alpha[w] / (denom[t * H + h] + 1e-16f) * scale;
  const float* hsrc = hbuf + (size_t)s * H * D + (size_t)h * D;
  float* od = concat ? (acc + (size_t)t * H * D + (size_t)h * D)
                     : (acc + (size_t)t * D);
  for (int d = lane; d < D; d += 32) atomicAdd(&od[d], hsrc[d] * wgt);
}

__global__ void k_elu_inplace(float* __restrict__ x, size_t n) {
  size_t i = (size_t)blockIdx.x * blockDim.x + threadIdx.x;
  if (i >= n) return;
  float v = x[i];
  x[i] = v > 0.f ? v : (__expf(v) - 1.f);
}

// =====================================================================
// host side
// =====================================================================
static inline unsigned nblk(size_t n, unsigned b) { return (unsigned)((n + b - 1) / b); }

static void launch_gemm(const unsigned short* A, const unsigned short* B, const float* bias,
                        float* C, int M, int Nc, int K, hipStream_t s) {
  dim3 grid((Nc + 255) / 256, M / 16);   // 4 waves x 64 cols per block
  k_gemm_bf16_wmma<<<grid, 128, 0, s>>>((const __bf16*)A, (const __bf16*)B, bias, C, M, Nc, K);
}

struct Ws {
  float *bufA, *bufB, *hbuf;
  unsigned short *actb, *wbf, *lbf;
  float *asrc, *adst, *mmax, *denom, *alpha, *biasc;
};

static void run_gat_layer(const float* in_act, int K,
                          const float* W, const float* L, const float* b, const float* lb,
                          const float* a_s, const float* a_d,
                          int H, int D, int concat, int apply_elu,
                          float* acc_out, float* hbuf, const Ws& ws,
                          const long long* ei, int E, int N, hipStream_t st) {
  const int HD = H * D;
  const int Ln = concat ? HD : D;            // linear-branch output width
  const size_t nin = (size_t)N * K;
  k_f32_to_bf16<<<nblk(nin, 256), 256, 0, st>>>(in_act, ws.actb, nin);
  k_f32_to_bf16<<<nblk((size_t)HD * K, 256), 256, 0, st>>>(W, ws.wbf, (size_t)HD * K);
  k_f32_to_bf16<<<nblk((size_t)Ln * K, 256), 256, 0, st>>>(L, ws.lbf, (size_t)Ln * K);
  k_add_vec<<<nblk(Ln, 256), 256, 0, st>>>(b, lb, ws.biasc, Ln);

  // GAT features h = act @ W^T   (bias folded into acc init instead)
  launch_gemm(ws.actb, ws.wbf, nullptr, hbuf, N, HD, K, st);
  // acc = act @ L^T + (lb + b)   (messages scatter-added on top)
  launch_gemm(ws.actb, ws.lbf, ws.biasc, acc_out, N, Ln, K, st);

  k_node_attn<<<nblk((size_t)N * H * 32, 256), 256, 0, st>>>(hbuf, a_s, a_d,
                                                             ws.asrc, ws.adst, N, H, D);
  k_init_softmax<<<nblk((size_t)N * H, 256), 256, 0, st>>>(ws.mmax, ws.denom, N * H);

  const size_t totE = (size_t)(E + N) * H;
  k_edge_pass1<<<nblk(totE, 256), 256, 0, st>>>(ei, E, N, H, ws.asrc, ws.adst, ws.alpha, ws.mmax);
  k_edge_pass2<<<nblk(totE, 256), 256, 0, st>>>(ei, E, N, H, ws.alpha, ws.mmax, ws.denom);
  k_edge_msg<<<nblk(totE * 32, 256), 256, 0, st>>>(ei, E, N, H, D, ws.alpha, ws.denom,
                                                   hbuf, acc_out, concat,
                                                   concat ? 1.0f : 1.0f / (float)H);
  if (apply_elu)
    k_elu_inplace<<<nblk((size_t)N * Ln, 256), 256, 0, st>>>(acc_out, (size_t)N * Ln);
}

extern "C" void kernel_launch(void* const* d_in, const int* in_sizes, int n_in,
                              void* d_out, int out_size, void* d_ws, size_t ws_size,
                              hipStream_t stream) {
  const float*     x  = (const float*)d_in[0];
  const long long* ei = (const long long*)d_in[1];          // int64 [2, E]
  const float *W1 = (const float*)d_in[2],  *as1 = (const float*)d_in[3],
              *ad1 = (const float*)d_in[4], *b1  = (const float*)d_in[5],
              *L1 = (const float*)d_in[6],  *lb1 = (const float*)d_in[7];
  const float *W2 = (const float*)d_in[8],  *as2 = (const float*)d_in[9],
              *ad2 = (const float*)d_in[10],*b2  = (const float*)d_in[11],
              *L2 = (const float*)d_in[12], *lb2 = (const float*)d_in[13];
  const float *W3 = (const float*)d_in[14], *as3 = (const float*)d_in[15],
              *ad3 = (const float*)d_in[16],*b3  = (const float*)d_in[17],
              *L3 = (const float*)d_in[18], *lb3 = (const float*)d_in[19];

  const int N = in_sizes[0] / 512;       // 50000
  const int E = in_sizes[1] / 2;         // 200000
  const size_t NB = (size_t)N * 1024;    // widest activation

  // ---- carve workspace ----
  Ws ws;
  float* f = (float*)d_ws;
  ws.bufA = f;            f += NB;
  ws.bufB = f;            f += NB;
  ws.hbuf = f;            f += NB;
  unsigned short* u = (unsigned short*)f;
  ws.actb = u;            u += NB;                 // bf16 activations
  ws.wbf  = u;            u += (size_t)1024 * 1024;
  ws.lbf  = u;            u += (size_t)1024 * 1024;
  f = (float*)u;
  ws.asrc  = f;           f += (size_t)N * 6;
  ws.adst  = f;           f += (size_t)N * 6;
  ws.mmax  = f;           f += (size_t)N * 6;
  ws.denom = f;           f += (size_t)N * 6;
  ws.alpha = f;           f += (size_t)(E + N) * 6;
  ws.biasc = f;           f += 1024;

  // layer1: 512 -> 4x256 concat, +ELU   (out: bufA)
  run_gat_layer(x, 512, W1, L1, b1, lb1, as1, ad1, 4, 256, 1, 1,
                ws.bufA, ws.hbuf, ws, ei, E, N, stream);
  // layer2: 1024 -> 4x256 concat, +ELU  (out: bufB)
  run_gat_layer(ws.bufA, 1024, W2, L2, b2, lb2, as2, ad2, 4, 256, 1, 1,
                ws.bufB, ws.hbuf, ws, ei, E, N, stream);
  // layer3: 1024 -> 6x64 mean, no ELU   (out: d_out)
  run_gat_layer(ws.bufB, 1024, W3, L3, b3, lb3, as3, ad3, 6, 64, 0, 0,
                (float*)d_out, ws.hbuf, ws, ei, E, N, stream);
}
